// GaussianRBF_61186104099013
// MI455X (gfx1250) — matrix operand order repaired and checked by reading, not compiled, not run on hardware
//
#include <hip/hip_runtime.h>

// Gaussian RBF: out[m,n] = exp(-0.5*(||x_m||^2 - 2 x_m.mu_n + ||mu_n||^2)*exp(-2 log_sigma_n))
//
// GEMM core: error-compensated bf16 split (x=xh+xl, mu=mh+ml) on
// V_WMMA_F32_16X16X32_BF16:  cross ~= xh*mh + xh*ml + xl*mh  (f32 accumulate).
// 6 bf16 WMMAs per 16x16 tile instead of 16 f32 WMMAs -> under the 64MB-store
// roofline (~3.1us @ 23.3TB/s) on the XDL bf16 rate, with TRANS co-execution
// against the f32 exp/norm epilogue. Norms and exp stay exact f32.

typedef __attribute__((ext_vector_type(2)))  float  v2f;
typedef __attribute__((ext_vector_type(4)))  float  v4f;
typedef __attribute__((ext_vector_type(8)))  float  v8f;
typedef __attribute__((ext_vector_type(16))) __bf16 v16bf;

namespace {
constexpr int kD = 64;               // K
constexpr int kO = 512;              // N
constexpr int kM = 8 * 4096;         // 32768 rows (B*S)
constexpr int kNtPerWave = 8;        // 8 n-tiles (128 cols) per wave
constexpr int kNGroups = kO / (16 * kNtPerWave);            // 4
constexpr int kMTiles = kM / 16;                            // 2048
constexpr int kWavesPerBlock = 8;                           // 256 threads
constexpr int kBlocks = (kMTiles * kNGroups) / kWavesPerBlock;  // 1024

// d_ws layout (all 32B-aligned)
constexpr size_t kWsMusH = 0;                        // 512*64 bf16 = 64 KB
constexpr size_t kWsMusL = kWsMusH + kO * kD * 2;    // 64 KB
constexpr size_t kWsM2   = kWsMusL + kO * kD * 2;    // 512 f32
constexpr size_t kWsInv2 = kWsM2 + kO * 4;           // 512 f32
}

// ---------------- Pre-pass: split mus into bf16 hi/lo, precompute ||mu||^2
// and exp(-2*log_sigma). One wave per mus row; trivial cost (128 KB touched).
__global__ __launch_bounds__(32)
void rbf_prep_kernel(const float* __restrict__ mus,
                     const float* __restrict__ log_sigmas,
                     __bf16* __restrict__ mus_h, __bf16* __restrict__ mus_l,
                     float* __restrict__ m2, float* __restrict__ inv2)
{
    const int n = blockIdx.x;
    const int lane = threadIdx.x;
    float s = 0.0f;
#pragma unroll
    for (int j = 0; j < 2; ++j) {
        const int d = lane + 32 * j;
        const float f = mus[(size_t)n * kD + d];
        s += f * f;
        const __bf16 h = (__bf16)f;
        const float  r = f - (float)h;
        mus_h[(size_t)n * kD + d] = h;
        mus_l[(size_t)n * kD + d] = (__bf16)r;
    }
    s += __shfl_xor(s, 16, 32);
    s += __shfl_xor(s, 8, 32);
    s += __shfl_xor(s, 4, 32);
    s += __shfl_xor(s, 2, 32);
    s += __shfl_xor(s, 1, 32);
    if (lane == 0) {
        m2[n]   = s;
        inv2[n] = __expf(-2.0f * log_sigmas[n]);
    }
}

// ---------------- Main kernel: one wave = 16-row m-tile x 8 n-tiles.
__global__ __launch_bounds__(256)
void GaussianRBF_wmma_kernel(const float* __restrict__ x,
                             const __bf16* __restrict__ mus_h,
                             const __bf16* __restrict__ mus_l,
                             const float* __restrict__ m2,
                             const float* __restrict__ inv2,
                             float* __restrict__ out)
{
    const int lane = threadIdx.x & 31;
    const int wave = threadIdx.x >> 5;
    const int w    = blockIdx.x * kWavesPerBlock + wave;

    const int mt = w / kNGroups;
    const int ng = w % kNGroups;
    const int m0 = mt * 16;
    const int n0base = ng * (16 * kNtPerWave);

    const int lm = lane & 15;
    const int hi = lane >> 4;

    // ---- Load A rows (f32) and build bf16 hi/lo fragments in-register.
    // 16-bit A 16x32 layout: lane holds row m0+lm, element e <-> K = 32ks + 16hi + e
    // (contiguous, matching B).
    const float* arow = x + (size_t)(m0 + lm) * kD;
    v16bf ah[2], al[2];
    float sA = 0.0f;
#pragma unroll
    for (int ks = 0; ks < 2; ++ks) {
        float f[16];
        const v4f q0 = *reinterpret_cast<const v4f*>(arow + ks * 32 + 16 * hi);
        const v4f q1 = *reinterpret_cast<const v4f*>(arow + ks * 32 + 16 * hi + 4);
        const v4f q2 = *reinterpret_cast<const v4f*>(arow + ks * 32 + 16 * hi + 8);
        const v4f q3 = *reinterpret_cast<const v4f*>(arow + ks * 32 + 16 * hi + 12);
#pragma unroll
        for (int j = 0; j < 4; ++j) { f[j] = q0[j]; f[4 + j] = q1[j]; f[8 + j] = q2[j]; f[12 + j] = q3[j]; }
#pragma unroll
        for (int e = 0; e < 16; ++e) {
            sA += f[e] * f[e];
            const __bf16 h = (__bf16)f[e];
            const float  r = f[e] - (float)h;
            ah[ks][e] = h;
            al[ks][e] = (__bf16)r;
        }
    }
    // lane L and L^16 together cover all K of row (L&15)
    const float rowx2 = sA + __shfl_xor(sA, 16, 32);
    // C layout: VGPR r holds row m0 + r + 8*hi
    float x2r[8];
#pragma unroll
    for (int r = 0; r < 8; ++r)
        x2r[r] = __shfl(rowx2, r + (hi << 3), 32);

    // ---- Loop over n-tiles; A fragments stay register-resident.
    for (int nt = 0; nt < kNtPerWave; ++nt) {
        const int n0 = n0base + nt * 16;
        const int n  = n0 + lm;

        // B fragment: element e <-> K = 32ks + e + 16hi (contiguous 16 bf16 = 32 bytes)
        const __bf16* bhp = mus_h + (size_t)n * kD + 16 * hi;
        const __bf16* blp = mus_l + (size_t)n * kD + 16 * hi;
        v16bf bh[2], bl[2];
#pragma unroll
        for (int ks = 0; ks < 2; ++ks) {
            bh[ks] = *reinterpret_cast<const v16bf*>(bhp + ks * 32);
            bl[ks] = *reinterpret_cast<const v16bf*>(blp + ks * 32);
        }

        v8f c = {};
#pragma unroll
        for (int ks = 0; ks < 2; ++ks) {
            // cross = xh*mh + xh*ml + xl*mh  (f32 accumulator carries the sum)
            c = __builtin_amdgcn_wmma_f32_16x16x32_bf16(
                    false, ah[ks], false, bh[ks], (short)0, c, false, false);
            c = __builtin_amdgcn_wmma_f32_16x16x32_bf16(
                    false, ah[ks], false, bl[ks], (short)0, c, false, false);
            c = __builtin_amdgcn_wmma_f32_16x16x32_bf16(
                    false, al[ks], false, bh[ks], (short)0, c, false, false);
        }

        const float m2n   = m2[n];
        const float inv2n = inv2[n];

        // Fused epilogue + non-temporal store (64 MB output, never re-read).
        float* ocol = out + (size_t)(m0 + 8 * hi) * kO + n;
#pragma unroll
        for (int r = 0; r < 8; ++r) {
            const float sq = x2r[r] - 2.0f * c[r] + m2n;
            const float v  = __expf(-0.5f * sq * inv2n);
            __builtin_nontemporal_store(v, ocol + (size_t)r * kO);
        }
    }
}

extern "C" void kernel_launch(void* const* d_in, const int* in_sizes, int n_in,
                              void* d_out, int out_size, void* d_ws, size_t ws_size,
                              hipStream_t stream) {
    (void)in_sizes; (void)n_in; (void)out_size; (void)ws_size;
    const float* x          = (const float*)d_in[0];   // (8, 4096, 64)
    const float* mus        = (const float*)d_in[1];   // (512, 64)
    const float* log_sigmas = (const float*)d_in[2];   // (512,)
    float* out              = (float*)d_out;           // (8, 4096, 512)

    char* ws = (char*)d_ws;
    __bf16* mus_h = (__bf16*)(ws + kWsMusH);
    __bf16* mus_l = (__bf16*)(ws + kWsMusL);
    float*  m2    = (float*)(ws + kWsM2);
    float*  inv2  = (float*)(ws + kWsInv2);

    rbf_prep_kernel<<<kO, 32, 0, stream>>>(mus, log_sigmas, mus_h, mus_l, m2, inv2);
    GaussianRBF_wmma_kernel<<<kBlocks, 256, 0, stream>>>(x, mus_h, mus_l, m2, inv2, out);
}
